// PraxisAttention_31980326486093
// MI455X (gfx1250) — matrix operand order repaired and checked by reading, not compiled, not run on hardware
//
#include <hip/hip_runtime.h>
#include <stdint.h>

typedef _Float16 f16;
typedef __attribute__((ext_vector_type(4)))  _Float16 v4h;
typedef __attribute__((ext_vector_type(8)))  _Float16 v8h;
typedef __attribute__((ext_vector_type(16))) _Float16 v16h;
typedef __attribute__((ext_vector_type(8)))  float    v8f;

static constexpr int S  = 2048;
static constexpr int D  = 1024;
static constexpr int H  = 16;
static constexpr int HD = 64;
static constexpr int BB = 2;
static constexpr int M  = BB * S;             // 4096 rows
static constexpr float LOG2E = 1.44269504088896340736f;

#define WMMA(A, B, C) \
  __builtin_amdgcn_wmma_f32_16x16x32_f16(false, (A), false, (B), (short)0, (C), false, false)

// Build a v16h fragment from two contiguous 16-byte chunks (global or LDS).
__device__ __forceinline__ v16h make_frag(const f16* p0, const f16* p1) {
  v8h lo = *(const v8h*)p0;
  v8h hi = *(const v8h*)p1;
  v16h r;
#pragma unroll
  for (int e = 0; e < 8; ++e) { r[e] = lo[e]; r[e + 8] = hi[e]; }
  return r;
}

// Async global -> LDS copy (CDNA5 VGLOBAL async op, tracked by ASYNCcnt).
__device__ __forceinline__ void async_copy_b128(uint32_t lds_off, const void* g) {
  asm volatile("global_load_async_to_lds_b128 %0, %1, off"
               :: "v"(lds_off), "v"((uint64_t)(uintptr_t)g)
               : "memory");
}
__device__ __forceinline__ void wait_async0() {
  asm volatile("s_wait_asynccnt 0x0" ::: "memory");
}
__device__ __forceinline__ void wait_async2() {   // leave 2 newest outstanding
  asm volatile("s_wait_asynccnt 0x2" ::: "memory");
}
// Generic LDS pointer -> wave-relative LDS byte offset (low 32 bits).
__device__ __forceinline__ uint32_t lds_off_of(const void* p) {
  return (uint32_t)(uintptr_t)p;
}

// ---- 16-lane butterfly reductions in pure VALU via DPP16 -------------------
template <int CTRL>
__device__ __forceinline__ float dpp_mov(float x) {
  return __builtin_bit_cast(float,
      __builtin_amdgcn_update_dpp(0, __builtin_bit_cast(int, x),
                                  CTRL, 0xf, 0xf, true));
}
__device__ __forceinline__ float red_max16(float x) {
  x = fmaxf(x, dpp_mov<0xB1>(x));    // quad_perm [1,0,3,2]  (xor 1)
  x = fmaxf(x, dpp_mov<0x4E>(x));    // quad_perm [2,3,0,1]  (xor 2)
  x = fmaxf(x, dpp_mov<0x140>(x));   // row_mirror           (xor 15)
  x = fmaxf(x, dpp_mov<0x141>(x));   // row_half_mirror      (xor 7)
  return x;
}
__device__ __forceinline__ float red_sum16(float x) {
  x += dpp_mov<0xB1>(x);
  x += dpp_mov<0x4E>(x);
  x += dpp_mov<0x140>(x);
  x += dpp_mov<0x141>(x);
  return x;
}

// ---------------------------------------------------------------------------
// f32 -> f16 conversion (x -> x16)
// ---------------------------------------------------------------------------
__global__ __launch_bounds__(256) void cvt_f32_to_f16(const float* __restrict__ in,
                                                      f16* __restrict__ out, int n4) {
  int i = blockIdx.x * 256 + threadIdx.x;
  if (i < n4) {
    float4 v = ((const float4*)in)[i];
    v4h o = { (f16)v.x, (f16)v.y, (f16)v.z, (f16)v.w };
    ((v4h*)out)[i] = o;
  }
}

// ---------------------------------------------------------------------------
// GEMM: C[M x D] = A16[M x D] (f16, row-major) @ W[D x D] (f32) (+bias)*scale
// MODE 0: f16 out, head layout     [B,H,S,HD]
// MODE 1: f16 out, head transposed [B,H,HD,S]   (for V)
// MODE 2: f32 out, row-major       [M x D]      (final output)
// ---------------------------------------------------------------------------
static constexpr int LDA_S = 40;  // f16 stride (32 + 8 pad)
static constexpr int LDB_S = 40;

template <int MODE>
__global__ __launch_bounds__(256)
void gemm_wmma(const f16* __restrict__ A, const float* __restrict__ W,
               const float* __restrict__ bias, float scale,
               void* __restrict__ Cout) {
  __shared__ f16 lA[128 * LDA_S];
  __shared__ f16 lB[64 * LDB_S];

  const int tid  = threadIdx.x;
  const int wave = tid >> 5;
  const int lane = tid & 31;
  const int half = lane >> 4;
  const int ln   = lane & 15;
  const int mw   = wave & 3;
  const int nw   = wave >> 2;
  const int mbase = blockIdx.y * 128;
  const int nbase = blockIdx.x * 64;

  // Per-thread staging slots
  const int arow = tid & 127, aseg = tid >> 7;            // 128 x 2 chunks(16 f16)
  const uint32_t aOff = lds_off_of(lA + arow * LDA_S + aseg * 16);

  v8f acc[2][2] = {};

  for (int kb = 0; kb < D; kb += 32) {
    // ---- stage A tile (f16) via async copies: 128 rows x 32 k ----
    {
      const f16* g = A + (size_t)(mbase + arow) * D + kb + aseg * 16;
      async_copy_b128(aOff, g);
      async_copy_b128(aOff + 16, g + 8);
    }
    // ---- stage W tile transposed (f32 -> f16): lB[n][k], 64 x 32 ----
    {
      int kk  = tid >> 3;        // 0..31
      int nn0 = (tid & 7) * 8;   // 0,8,...,56
      const float4* wp = (const float4*)(W + (size_t)(kb + kk) * D + nbase + nn0);
      float4 w0 = wp[0], w1 = wp[1];
      float wv[8] = { w0.x, w0.y, w0.z, w0.w, w1.x, w1.y, w1.z, w1.w };
#pragma unroll
      for (int j = 0; j < 8; ++j) lB[(nn0 + j) * LDB_S + kk] = (f16)wv[j];
    }
    wait_async0();
    __syncthreads();

    v16h af[2], bf[2];
#pragma unroll
    for (int ms = 0; ms < 2; ++ms) {
      const f16* r = lA + (mw * 32 + ms * 16 + ln) * LDA_S;
      af[ms] = make_frag(r + half * 16, r + half * 16 + 8);
    }
#pragma unroll
    for (int ns = 0; ns < 2; ++ns) {
      const f16* r = lB + (nw * 32 + ns * 16 + ln) * LDB_S + half * 16;
      bf[ns] = make_frag(r, r + 8);
    }
#pragma unroll
    for (int ms = 0; ms < 2; ++ms)
#pragma unroll
      for (int ns = 0; ns < 2; ++ns)
        acc[ms][ns] = WMMA(af[ms], bf[ns], acc[ms][ns]);

    __syncthreads();
  }

  // ---- epilogue ----
#pragma unroll
  for (int ms = 0; ms < 2; ++ms)
#pragma unroll
    for (int ns = 0; ns < 2; ++ns)
#pragma unroll
      for (int r = 0; r < 8; ++r) {
        int row = mbase + mw * 32 + ms * 16 + half * 8 + r;
        int col = nbase + nw * 32 + ns * 16 + ln;
        float v = acc[ms][ns][r];
        if (bias) v += bias[col];
        v *= scale;
        if constexpr (MODE == 0) {
          int b = row >> 11, s = row & (S - 1);
          int h = col >> 6,  d = col & (HD - 1);
          ((f16*)Cout)[((((size_t)b * H + h) * S) + s) * HD + d] = (f16)v;
        } else if constexpr (MODE == 1) {
          int b = row >> 11, s = row & (S - 1);
          int h = col >> 6,  d = col & (HD - 1);
          ((f16*)Cout)[((((size_t)b * H + h) * HD) + d) * S + s] = (f16)v;
        } else {
          ((float*)Cout)[(size_t)row * D + col] = v;
        }
      }
}

// ---------------------------------------------------------------------------
// Flash attention: grid (S/128, B*H), 8 waves/block, each wave owns 16 query
// rows. Q pre-scaled by log2e/sqrt(HD). K/V tiles double-buffered in LDS via
// async global->LDS copies (copy t+1 overlaps compute t). ALiBi + causal mask
// + online softmax in base-2 with DPP16 butterfly reductions.
// ---------------------------------------------------------------------------
static constexpr int PSTR = 40;  // per-wave P scratch row stride (32 + 8 pad)
static constexpr int KSTR = 72;  // K tile row stride (64 + 8 pad), 144B = 9*16
static constexpr int VSTR = 40;  // V tile row stride (32 + 8 pad), 80B = 5*16

__global__ __launch_bounds__(256)
void flash_attn(const f16* __restrict__ q16, const f16* __restrict__ k16,
                const f16* __restrict__ vT16, f16* __restrict__ ctx) {
  __shared__ f16 kS[2][32 * KSTR];       // [buf][key][dim]
  __shared__ f16 vS[2][64 * VSTR];       // [buf][dim][key]
  __shared__ f16 pS[8 * 16 * PSTR];      // per-wave P relayout scratch

  const int tid  = threadIdx.x;
  const int wave = tid >> 5;
  const int lane = tid & 31;
  const int half = lane >> 4;
  const int ln   = lane & 15;

  const int bh = blockIdx.y;
  const int b  = bh >> 4;
  const int h  = bh & 15;
  const int qblock = blockIdx.x * 128;
  const int qbase  = qblock + wave * 16;

  const f16* qp = q16 + (size_t)bh * S * HD;
  const f16* kp = k16 + (size_t)bh * S * HD;
  const f16* vp = vT16 + (size_t)bh * HD * S;
  f16* pbuf = pS + wave * (16 * PSTR);

  // ALiBi slope in base-2 domain (scores already carry log2e from Q scaling)
  const float slope2 = exp2f(-0.5f * (float)(h + 1)) * LOG2E;

  // Q fragments (16 rows x 64 dims -> two 16x32 A fragments), loaded once.
  const f16* qrow = qp + (size_t)(qbase + ln) * HD;
  v16h qf0 = make_frag(qrow + half * 16,      qrow + half * 16 + 8);
  v16h qf1 = make_frag(qrow + 32 + half * 16, qrow + 40 + half * 16);

  // Per-thread staging slots (constant across tiles).
  const int krow = tid >> 3, kseg = tid & 7;   // 32 rows x 8 chunks of 16B
  const int vrow = tid >> 2, vseg = tid & 3;   // 64 rows x 4 chunks of 16B
  uint32_t kOff[2], vOff[2];
#pragma unroll
  for (int u = 0; u < 2; ++u) {
    kOff[u] = lds_off_of(&kS[u][krow * KSTR + kseg * 8]);
    vOff[u] = lds_off_of(&vS[u][vrow * VSTR + vseg * 8]);
  }
  const f16* kg = kp + (size_t)krow * HD + kseg * 8;  // + k0*HD per tile
  const f16* vg = vp + (size_t)vrow * S;              // + k0 per tile

  float mx[8], l[8];
  v8f o[4] = {};
#pragma unroll
  for (int r = 0; r < 8; ++r) { mx[r] = -3.0e38f; l[r] = 0.0f; }

  const int ktiles = qblock / 32 + 4;  // block-uniform causal trip count

  // Prologue: stage tile 0 into buffer 0.
  async_copy_b128(kOff[0], kg);
  async_copy_b128(vOff[0], vg);

  for (int kt = 0; kt < ktiles; ++kt) {
    const int k0 = kt * 32;
    const int buf = kt & 1;

    // Stage tile kt+1 into the other buffer, then wait for tile kt only
    // (ASYNCcnt completes in order; 2 newest may remain outstanding).
    if (kt + 1 < ktiles) {
      async_copy_b128(kOff[buf ^ 1], kg + (size_t)(k0 + 32) * HD);
      async_copy_b128(vOff[buf ^ 1], vg + k0 + 32);
      wait_async2();
    } else {
      wait_async0();
    }
    __syncthreads();   // all waves' copies for tile kt have landed

    // ---- scores: 16q x 32k, two 16x16 accumulators ----
    const f16* kbase = kS[buf];
    v8f sc0 = {}, sc1 = {};
    {
      const f16* kr = kbase + ln * KSTR + half * 16;
      sc0 = WMMA(qf0, make_frag(kr, kr + 8), sc0);
      sc0 = WMMA(qf1, make_frag(kr + 32, kr + 40), sc0);
      const f16* kr1 = kbase + (16 + ln) * KSTR + half * 16;
      sc1 = WMMA(qf0, make_frag(kr1, kr1 + 8), sc1);
      sc1 = WMMA(qf1, make_frag(kr1 + 32, kr1 + 40), sc1);
    }

    // ---- ALiBi + causal mask + online softmax (base-2, DPP reductions) ----
    float p0a[8], p1a[8], corr[8];
    const int key0 = k0 + ln, key1 = k0 + 16 + ln;
#pragma unroll
    for (int r = 0; r < 8; ++r) {
      int qi = qbase + half * 8 + r;
      float v0 = sc0[r] - slope2 * (float)key0;
      float v1 = sc1[r] - slope2 * (float)key1;
      if (key0 > qi) v0 = -1.0e30f;
      if (key1 > qi) v1 = -1.0e30f;
      float t = red_max16(fmaxf(v0, v1));
      float mnew = fmaxf(mx[r], t);
      float c  = exp2f(mx[r] - mnew);
      float e0 = exp2f(v0 - mnew);
      float e1 = exp2f(v1 - mnew);
      float rs = red_sum16(e0 + e1);
      l[r] = l[r] * c + rs;
      mx[r] = mnew;
      corr[r] = c;
      p0a[r] = e0;
      p1a[r] = e1;
    }
#pragma unroll
    for (int cc = 0; cc < 4; ++cc)
#pragma unroll
      for (int r = 0; r < 8; ++r) o[cc][r] *= corr[r];

    // ---- relayout P (C-frag -> A-frag) via per-wave LDS scratch ----
#pragma unroll
    for (int r = 0; r < 8; ++r) {
      pbuf[(half * 8 + r) * PSTR + ln]      = (f16)p0a[r];
      pbuf[(half * 8 + r) * PSTR + 16 + ln] = (f16)p1a[r];
    }
    asm volatile("s_wait_dscnt 0x0" ::: "memory");
    v16h pa = make_frag(pbuf + ln * PSTR + half * 16,
                        pbuf + ln * PSTR + half * 16 + 8);

    // ---- O += P @ V ----
#pragma unroll
    for (int cc = 0; cc < 4; ++cc) {
      const f16* vc = vS[buf] + (cc * 16 + ln) * VSTR + half * 16;
      o[cc] = WMMA(pa, make_frag(vc, vc + 8), o[cc]);
    }

    __syncthreads();  // all waves done reading buf before it is re-staged
  }

  // ---- normalize and store ctx [B,S,H*HD] f16 ----
#pragma unroll
  for (int cc = 0; cc < 4; ++cc)
#pragma unroll
    for (int r = 0; r < 8; ++r) {
      int qi  = qbase + half * 8 + r;
      int col = h * HD + cc * 16 + ln;
      ctx[(size_t)(b * S + qi) * D + col] = (f16)(o[cc][r] / l[r]);
    }
}

// ---------------------------------------------------------------------------
extern "C" void kernel_launch(void* const* d_in, const int* in_sizes, int n_in,
                              void* d_out, int out_size, void* d_ws, size_t ws_size,
                              hipStream_t stream) {
  (void)in_sizes; (void)n_in; (void)out_size; (void)ws_size;
  const float* x  = (const float*)d_in[0];
  const float* Wq = (const float*)d_in[1];
  const float* bq = (const float*)d_in[2];
  const float* Wk = (const float*)d_in[3];
  const float* Wv = (const float*)d_in[4];
  const float* Wo = (const float*)d_in[5];
  const float* bo = (const float*)d_in[6];

  const size_t NE = (size_t)M * D;  // 4 Mi elements per buffer
  f16* x16  = (f16*)d_ws;
  f16* q16  = x16 + NE;
  f16* k16  = q16 + NE;
  f16* vT16 = k16 + NE;
  f16* ctx  = vT16 + NE;

  // 1) x -> f16
  cvt_f32_to_f16<<<dim3((unsigned)(NE / 4 / 256)), dim3(256), 0, stream>>>(x, x16, (int)(NE / 4));

  dim3 ggrid(D / 64, M / 128);   // (16, 32)
  dim3 blk(256);

  // 2) projections; Q carries 1/sqrt(HD) * log2e for base-2 softmax
  gemm_wmma<0><<<ggrid, blk, 0, stream>>>(x16, Wq, bq,      0.125f * LOG2E, (void*)q16);
  gemm_wmma<0><<<ggrid, blk, 0, stream>>>(x16, Wk, nullptr, 1.0f,           (void*)k16);
  gemm_wmma<1><<<ggrid, blk, 0, stream>>>(x16, Wv, nullptr, 1.0f,           (void*)vT16);

  // 3) attention
  dim3 agrid(S / 128, BB * H);   // (16, 32)
  flash_attn<<<agrid, blk, 0, stream>>>(q16, k16, vT16, ctx);

  // 4) output projection -> f32
  gemm_wmma<2><<<ggrid, blk, 0, stream>>>(ctx, Wo, bo, 1.0f, d_out);
}
